// SAGE_GCN_30915174597107
// MI455X (gfx1250) — compile-verified, matrix operand order checked by reading, not run on hardware
//
#include <hip/hip_runtime.h>
#include <hip/hip_bf16.h>
#include <math.h>

typedef _Float16 half_t;
typedef __attribute__((ext_vector_type(16))) _Float16 v16h;
typedef __attribute__((ext_vector_type(8)))  _Float16 v8h;
typedef __attribute__((ext_vector_type(8)))  float    v8f;

#define H_DIM 64
#define GEMM_MT 2   // 16-row tiles per wave
#define GEMM_NT 4   // 16-col tiles per wave  -> 32x64 output strip / wave

// ---------------- helpers ----------------
__device__ __forceinline__ int enc_ord(float f) {
    int i = __float_as_int(f);
    return i >= 0 ? i : (i ^ 0x7FFFFFFF);
}
__device__ __forceinline__ float dec_ord(int i) {
    return i >= 0 ? __int_as_float(i) : __int_as_float(i ^ 0x7FFFFFFF);
}

// ---------------- conversion / packing ----------------
__global__ void k_f32_to_f16(const float* __restrict__ in, half_t* __restrict__ out, int n) {
    int i = blockIdx.x * blockDim.x + threadIdx.x;
    if (i < n) out[i] = (half_t)in[i];
}

// out[H][K1+K2] = [wl | wr] converted to f16
__global__ void k_pack_w(const float* __restrict__ wl, const float* __restrict__ wr,
                         half_t* __restrict__ out, int Hh, int K1, int K2) {
    int KT = K1 + K2;
    int i = blockIdx.x * blockDim.x + threadIdx.x;
    if (i >= Hh * KT) return;
    int r = i / KT, k = i % KT;
    float v = (k < K1) ? wl[r * K1 + k] : wr[r * K2 + (k - K1)];
    out[i] = (half_t)v;
}

__global__ void k_fill_i32(int* __restrict__ p, int val, int n) {
    int i = blockIdx.x * blockDim.x + threadIdx.x;
    if (i < n) p[i] = val;
}

// ---------------- WMMA GEMM ----------------
// C[M,Ncols] = act( A[M,K](f16,row-major) * Bw[Ncols,K](f16,row-major)^T + bias )
// One 32x64 strip per wave32 (2x4 accumulators): 8 WMMAs per 12 b128 loads.
// K multiple of 32, Ncols multiple of 64. act: 0=none, 1=sigmoid, 2=relu.
__global__ void k_wmma_gemm(const half_t* __restrict__ A, const half_t* __restrict__ Bw,
                            const float* __restrict__ bias,
                            float* __restrict__ out32, half_t* __restrict__ out16,
                            int M, int Ncols, int K, int act) {
    int lane = threadIdx.x & 31;
    int wave = threadIdx.x >> 5;
    int nstrips = Ncols >> 6;                 // 64-wide col strips
    int mstrips = (M + 31) >> 5;              // 32-high row strips
    int strip = blockIdx.x * (blockDim.x >> 5) + wave;
    if (strip >= mstrips * nstrips) return;
    int ms = strip / nstrips, ns = strip % nstrips;

    // A 16-bit 16x32 layout: lanes 0-15 hold K 0..7 (v0-3) and 16..23 (v4-7),
    // lanes 16-31 hold K 8..15 and 24..31; row = lane&15. B mirrors with col = lane&15.
    int kofs = (lane >> 4) * 8;
    const half_t* pa[GEMM_MT];
    const half_t* pb[GEMM_NT];
#pragma unroll
    for (int i = 0; i < GEMM_MT; ++i) {
        int r = ms * 32 + i * 16 + (lane & 15);
        if (r >= M) r = M - 1;                // clamp read (writes guarded below)
        pa[i] = A + (size_t)r * K + kofs;
    }
#pragma unroll
    for (int t = 0; t < GEMM_NT; ++t) {
        int c = ns * 64 + t * 16 + (lane & 15);
        pb[t] = Bw + (size_t)c * K + kofs;
    }

    v8f acc[GEMM_MT][GEMM_NT] = {};
    for (int k0 = 0; k0 < K; k0 += 32) {
        v16h av[GEMM_MT], bv[GEMM_NT];
#pragma unroll
        for (int i = 0; i < GEMM_MT; ++i) {
            v8h a0 = *(const v8h*)(pa[i] + k0);
            v8h a1 = *(const v8h*)(pa[i] + k0 + 16);
#pragma unroll
            for (int j = 0; j < 8; ++j) { av[i][j] = a0[j]; av[i][8 + j] = a1[j]; }
        }
#pragma unroll
        for (int t = 0; t < GEMM_NT; ++t) {
            v8h b0 = *(const v8h*)(pb[t] + k0);
            v8h b1 = *(const v8h*)(pb[t] + k0 + 16);
#pragma unroll
            for (int j = 0; j < 8; ++j) { bv[t][j] = b0[j]; bv[t][8 + j] = b1[j]; }
        }
#pragma unroll
        for (int i = 0; i < GEMM_MT; ++i)
#pragma unroll
            for (int t = 0; t < GEMM_NT; ++t)
                acc[i][t] = __builtin_amdgcn_wmma_f32_16x16x32_f16(
                    false, av[i], false, bv[t], (short)0, acc[i][t], false, false);
    }

#pragma unroll
    for (int i = 0; i < GEMM_MT; ++i) {
        int rbase = ms * 32 + i * 16 + ((lane >> 4) * 8);  // D layout: vgpr r -> M = rbase + r
#pragma unroll
        for (int t = 0; t < GEMM_NT; ++t) {
            int col = ns * 64 + t * 16 + (lane & 15);
            float bc = bias ? bias[col] : 0.0f;
#pragma unroll
            for (int r = 0; r < 8; ++r) {
                int orow = rbase + r;
                if (orow >= M) continue;
                float v = acc[i][t][r] + bc;
                if (act == 1)      v = 1.0f / (1.0f + __expf(-v));
                else if (act == 2) v = fmaxf(v, 0.0f);
                size_t o = (size_t)orow * Ncols + col;
                if (out32) out32[o] = v;
                if (out16) out16[o] = (half_t)v;
            }
        }
    }
}

// ---------------- SAGE aggregation ----------------
// one thread per (edge, feature). Gathers hit L2 (25.6MB table << 192MB L2).
__global__ void k_edge(const int* __restrict__ ei, const float* __restrict__ h,
                       float* __restrict__ s, float* __restrict__ s2,
                       int* __restrict__ mx, float* __restrict__ cnt, int E) {
    int idx = blockIdx.x * blockDim.x + threadIdx.x;
    if (idx >= E * H_DIM) return;
    int e = idx >> 6, f = idx & 63;
    int src = ei[e];
    int dst = ei[E + e];
    float v = h[(size_t)src * H_DIM + f];
    size_t o = (size_t)dst * H_DIM + f;
    atomicAdd(&s[o], v);
    atomicAdd(&s2[o], v * v);
    atomicMax(&mx[o], enc_ord(v));
    if (f == 0) atomicAdd(&cnt[dst], 1.0f);
}

// feats[n][384] = [mean | max | sum | std | var | root] (f16)
__global__ void k_combine(const float* __restrict__ h, const float* __restrict__ s,
                          const float* __restrict__ s2, const int* __restrict__ mx,
                          const float* __restrict__ cnt, half_t* __restrict__ feats, int Nn) {
    int idx = blockIdx.x * blockDim.x + threadIdx.x;
    if (idx >= Nn * H_DIM) return;
    int n = idx >> 6, f = idx & 63;
    float c  = cnt[n];
    float cc = fmaxf(c, 1.0f);
    float sv = s[idx], qv = s2[idx];
    float mean = sv / cc;
    float var  = qv / cc - mean * mean;
    float stdv = sqrtf(fmaxf(var, 0.0f) + 1e-5f);
    float mv   = (c > 0.0f) ? dec_ord(mx[idx]) : 0.0f;  // empty segments -> 0 (PyG)
    half_t* fr = feats + (size_t)n * 384;
    fr[f]         = (half_t)mean;
    fr[64 + f]    = (half_t)mv;
    fr[128 + f]   = (half_t)sv;
    fr[192 + f]   = (half_t)stdv;
    fr[256 + f]   = (half_t)var;
    fr[320 + f]   = (half_t)h[idx];
}

// in-place row L2-normalize of x[N,64]: one wave32 per row, 2 elems/lane
__global__ void k_rownorm(float* __restrict__ x, int Nn) {
    int gid = blockIdx.x * blockDim.x + threadIdx.x;
    int row = gid >> 5, lane = gid & 31;
    if (row >= Nn) return;
    float* p = x + (size_t)row * H_DIM;
    float v0 = p[lane], v1 = p[lane + 32];
    float ss = v0 * v0 + v1 * v1;
#pragma unroll
    for (int o = 16; o > 0; o >>= 1) ss += __shfl_xor(ss, o, 32);
    float inv = 1.0f / fmaxf(sqrtf(ss), 1e-12f);
    p[lane] = v0 * inv;
    p[lane + 32] = v1 * inv;
}

__global__ void k_bn_stats(const float* __restrict__ x, float* __restrict__ stats, int Nn) {
    int gid = blockIdx.x * blockDim.x + threadIdx.x;
    int f = gid & 63;
    int r0 = gid >> 6;
    int stride = (int)((gridDim.x * blockDim.x) >> 6);
    float s = 0.0f, q = 0.0f;
    for (int r = r0; r < Nn; r += stride) {
        float v = x[(size_t)r * H_DIM + f];
        s += v; q += v * v;
    }
    atomicAdd(&stats[f], s);
    atomicAdd(&stats[64 + f], q);
}

// y = relu(BN(x)); writes f32 features + f16 slice of concat buffer hc[n][128]
__global__ void k_bn_apply(const float* __restrict__ in, const float* __restrict__ stats,
                           const float* __restrict__ g, const float* __restrict__ b,
                           float* __restrict__ out, half_t* __restrict__ hc, int hcoff, int Nn) {
    int idx = blockIdx.x * blockDim.x + threadIdx.x;
    if (idx >= Nn * H_DIM) return;
    int f = idx & 63, n = idx >> 6;
    float m   = stats[f] / (float)Nn;
    float var = stats[64 + f] / (float)Nn - m * m;     // biased, as torch BN
    float v = (in[idx] - m) * rsqrtf(var + 1e-5f) * g[f] + b[f];
    v = fmaxf(v, 0.0f);
    out[idx] = v;
    hc[(size_t)n * 128 + hcoff + f] = (half_t)v;
}

// ---------------- pooling + readout ----------------
__global__ void k_pool(const float* __restrict__ hc, const int* __restrict__ batch,
                       float* __restrict__ gsum, float* __restrict__ gcnt, int Nn) {
    int idx = blockIdx.x * blockDim.x + threadIdx.x;
    if (idx >= Nn * H_DIM) return;
    int n = idx >> 6, f = idx & 63;
    int b = batch[n];
    atomicAdd(&gsum[b * H_DIM + f], hc[idx]);
    if (f == 0) atomicAdd(&gcnt[b], 1.0f);
}

__global__ void k_final(const float* __restrict__ gsum, const float* __restrict__ gcnt,
                        const float* __restrict__ il_w, const float* __restrict__ il_b,
                        const float* __restrict__ hl1_w, const float* __restrict__ hl1_b,
                        const float* __restrict__ hl2_w, const float* __restrict__ hl2_b,
                        const float* __restrict__ ol_w, const float* __restrict__ ol_b,
                        float* __restrict__ out) {
    int g = threadIdx.x;
    if (g >= 64) return;
    float gv[64], o1[64], h1[16], h2[16];
    float c = fmaxf(gcnt[g], 1.0f);
    for (int k = 0; k < 64; ++k) gv[k] = gsum[g * 64 + k] / c;
    for (int j = 0; j < 64; ++j) {
        float a = il_b[j];
        for (int k = 0; k < 64; ++k) a += il_w[j * 64 + k] * gv[k];
        o1[j] = 1.0f / (1.0f + __expf(-a));
    }
    for (int j = 0; j < 16; ++j) {
        float a = hl1_b[j];
        for (int k = 0; k < 64; ++k) a += hl1_w[j * 64 + k] * o1[k];
        h1[j] = fmaxf(a, 0.0f);
    }
    for (int j = 0; j < 16; ++j) {
        float a = hl2_b[j];
        for (int k = 0; k < 16; ++k) a += hl2_w[j * 16 + k] * h1[k];
        h2[j] = fmaxf(a, 0.0f);
    }
    float a = ol_b[0];
    for (int k = 0; k < 16; ++k) a += ol_w[k] * h2[k];
    out[g] = a;
}

// ---------------- host orchestration ----------------
static inline int cdiv(long long a, long long b) { return (int)((a + b - 1) / b); }

extern "C" void kernel_launch(void* const* d_in, const int* in_sizes, int n_in,
                              void* d_out, int out_size, void* d_ws, size_t ws_size,
                              hipStream_t stream) {
    const float* x      = (const float*)d_in[0];
    const int*   ei     = (const int*)d_in[1];
    const int*   batch  = (const int*)d_in[2];
    const float* lin1_w = (const float*)d_in[3];
    const float* lin1_b = (const float*)d_in[4];
    const float* lin2_w = (const float*)d_in[5];
    const float* lin2_b = (const float*)d_in[6];
    const float* c1_wl  = (const float*)d_in[7];
    const float* c1_bl  = (const float*)d_in[8];
    const float* c1_wr  = (const float*)d_in[9];
    const float* bn1_g  = (const float*)d_in[10];
    const float* bn1_b  = (const float*)d_in[11];
    const float* c2_wl  = (const float*)d_in[12];
    const float* c2_bl  = (const float*)d_in[13];
    const float* c2_wr  = (const float*)d_in[14];
    const float* bn2_g  = (const float*)d_in[15];
    const float* bn2_b  = (const float*)d_in[16];
    const float* lin_w  = (const float*)d_in[17];
    const float* lin_b  = (const float*)d_in[18];
    const float* il_w   = (const float*)d_in[19];
    const float* il_b   = (const float*)d_in[20];
    const float* hl1_w  = (const float*)d_in[21];
    const float* hl1_b  = (const float*)d_in[22];
    const float* hl2_w  = (const float*)d_in[23];
    const float* hl2_b  = (const float*)d_in[24];
    const float* ol_w   = (const float*)d_in[25];
    const float* ol_b   = (const float*)d_in[26];

    const int Nn = in_sizes[0] / 128;   // 100000
    const int E  = in_sizes[1] / 2;     // 1600000

    // ---- workspace layout (256B-aligned regions) ----
    char* base = (char*)d_ws;
    size_t off = 0;
    auto alloc = [&](size_t bytes) -> char* {
        char* p = base + off;
        off = (off + bytes + 255) & ~(size_t)255;
        return p;
    };
    half_t* x16      = (half_t*)alloc((size_t)Nn * 128 * 2);
    half_t* w1_16    = (half_t*)alloc((size_t)512 * 128 * 2);
    half_t* w2_16    = (half_t*)alloc((size_t)64 * 512 * 2);
    half_t* wpack1   = (half_t*)alloc((size_t)64 * 384 * 2);
    half_t* wpack2   = (half_t*)alloc((size_t)64 * 384 * 2);
    half_t* wlin16   = (half_t*)alloc((size_t)64 * 128 * 2);
    half_t* h512_16  = (half_t*)alloc((size_t)Nn * 512 * 2);   // also aliased as feats16
    half_t* hc16     = (half_t*)alloc((size_t)Nn * 128 * 2);
    float*  h64      = (float*)alloc((size_t)Nn * 64 * 4);
    float*  x1       = (float*)alloc((size_t)Nn * 64 * 4);
    float*  x2       = (float*)alloc((size_t)Nn * 64 * 4);
    float*  s        = (float*)alloc((size_t)Nn * 64 * 4);
    float*  s2       = (float*)alloc((size_t)Nn * 64 * 4);
    int*    mx       = (int*)alloc((size_t)Nn * 64 * 4);
    float*  cnt      = (float*)alloc((size_t)Nn * 4);
    float*  tmp      = (float*)alloc((size_t)Nn * 64 * 4);
    float*  stats    = (float*)alloc(128 * 4);
    float*  gsum     = (float*)alloc(64 * 64 * 4);
    float*  gcnt     = (float*)alloc(64 * 4);
    half_t* feats16  = h512_16;   // alias: h512 consumed before feats is built
    (void)ws_size; (void)n_in; (void)out_size;

    const int TB = 256;
    // ---- precision conversion / weight packing ----
    k_f32_to_f16<<<cdiv((long long)Nn * 128, TB), TB, 0, stream>>>(x, x16, Nn * 128);
    k_f32_to_f16<<<cdiv(512 * 128, TB), TB, 0, stream>>>(lin1_w, w1_16, 512 * 128);
    k_f32_to_f16<<<cdiv(64 * 512, TB), TB, 0, stream>>>(lin2_w, w2_16, 64 * 512);
    k_f32_to_f16<<<cdiv(64 * 128, TB), TB, 0, stream>>>(lin_w, wlin16, 64 * 128);
    k_pack_w<<<cdiv(64 * 384, TB), TB, 0, stream>>>(c1_wl, c1_wr, wpack1, 64, 320, 64);
    k_pack_w<<<cdiv(64 * 384, TB), TB, 0, stream>>>(c2_wl, c2_wr, wpack2, 64, 320, 64);

    auto gemm = [&](const half_t* A, const half_t* Bw, const float* bias,
                    float* o32, half_t* o16, int M, int Ncols, int K, int act) {
        int strips = ((M + 31) / 32) * (Ncols / 64);
        k_wmma_gemm<<<cdiv(strips, 4), 128, 0, stream>>>(A, Bw, bias, o32, o16, M, Ncols, K, act);
    };

    // ---- MLP front end: h = relu(lin2(sigmoid(lin1(x)))) ----
    gemm(x16, w1_16, lin1_b, nullptr, h512_16, Nn, 512, 128, /*sigmoid*/1);
    gemm(h512_16, w2_16, lin2_b, h64, nullptr, Nn, 64, 512, /*relu*/2);

    // ---- SAGE conv (shared sequence) ----
    auto run_conv = [&](const float* hin, const half_t* wpack, const float* bl,
                        const float* bng, const float* bnb, float* xout, int hcoff) {
        (void)hipMemsetAsync(s,  0, (size_t)Nn * 64 * 4, stream);
        (void)hipMemsetAsync(s2, 0, (size_t)Nn * 64 * 4, stream);
        (void)hipMemsetAsync(cnt, 0, (size_t)Nn * 4, stream);
        k_fill_i32<<<cdiv((long long)Nn * 64, TB), TB, 0, stream>>>(mx, 0x807FFFFF, Nn * 64); // enc(-inf)
        k_edge<<<cdiv((long long)E * 64, TB), TB, 0, stream>>>(ei, hin, s, s2, mx, cnt, E);
        k_combine<<<cdiv((long long)Nn * 64, TB), TB, 0, stream>>>(hin, s, s2, mx, cnt, feats16, Nn);
        gemm(feats16, wpack, bl, tmp, nullptr, Nn, 64, 384, /*none*/0);
        k_rownorm<<<cdiv((long long)Nn * 32, TB), TB, 0, stream>>>(tmp, Nn);
        (void)hipMemsetAsync(stats, 0, 128 * 4, stream);
        k_bn_stats<<<512, TB, 0, stream>>>(tmp, stats, Nn);
        k_bn_apply<<<cdiv((long long)Nn * 64, TB), TB, 0, stream>>>(tmp, stats, bng, bnb, xout, hc16, hcoff, Nn);
    };

    run_conv(h64, wpack1, c1_bl, bn1_g, bn1_b, x1, 0);
    run_conv(x1,  wpack2, c2_bl, bn2_g, bn2_b, x2, 64);

    // ---- hc = relu(lin([x1|x2])) ----
    gemm(hc16, wlin16, lin_b, tmp, nullptr, Nn, 64, 128, /*relu*/2);

    // ---- global mean pool + readout MLP ----
    (void)hipMemsetAsync(gsum, 0, 64 * 64 * 4, stream);
    (void)hipMemsetAsync(gcnt, 0, 64 * 4, stream);
    k_pool<<<cdiv((long long)Nn * 64, TB), TB, 0, stream>>>(tmp, batch, gsum, gcnt, Nn);
    k_final<<<1, 64, 0, stream>>>(gsum, gcnt, il_w, il_b, hl1_w, hl1_b,
                                  hl2_w, hl2_b, ol_w, ol_b, (float*)d_out);
}
